// DistancePositionOperator_34883724378528
// MI455X (gfx1250) — matrix operand order repaired and checked by reading, compiled, not run on hardware
//
#include <hip/hip_runtime.h>

// CDNA5 / gfx1250 — wave32, WMMA 16x16x32 f16->f32,
// double-buffered TDM (TENSORcnt) j-tile streaming + async-LDS (ASYNCcnt) i-tile.
typedef __attribute__((ext_vector_type(16))) _Float16 v16h;
typedef __attribute__((ext_vector_type(8)))  float    v8f;
typedef unsigned int u32x4 __attribute__((ext_vector_type(4)));
typedef unsigned int u32x8 __attribute__((ext_vector_type(8)));

namespace {
constexpr int kB = 2;
constexpr int kL = 1024;
constexpr int kD = 128;
constexpr int kT = 16;              // tile edge
constexpr int kNJT = kL / kT;       // 64 j-tiles
constexpr int kThreads = 256;       // 8 waves; wave w owns D-chunk [16w,16w+16)
}

// Issue one TDM DMA of a contiguous 8KB (2048 x f32) j-tile into LDS.
__device__ __forceinline__ void tdm_load_tile(unsigned ldsAddr,
                                              unsigned long long ga) {
  constexpr unsigned nElem = kT * kD;                  // 2048 elements
  // D# group 0 (128b): count=1 | lds_addr | global_addr[56:0] | type=2
  u32x4 g0;
  g0[0] = 1u;                                          // count=1, user mode
  g0[1] = ldsAddr;                                     // lds_addr (bytes)
  g0[2] = (unsigned)(ga & 0xFFFFFFFFull);              // global_addr[31:0]
  g0[3] = (unsigned)((ga >> 32) & 0x1FFFFFFull)        // global_addr[56:32]
          | (2u << 30);                                // type=2 ("image")
  // D# group 1 (256b): data_size=4B, one row of nElem, stride=nElem
  u32x8 g1;
  g1[0] = (2u << 16);                // workgroup_mask=0, data_size=2 (4B)
  g1[1] = (nElem & 0xFFFFu) << 16;   // tensor_dim0[15:0]
  g1[2] = (nElem >> 16)              // tensor_dim0[31:16]
          | (1u << 16);              // tensor_dim1[15:0] = 1
  g1[3] = (nElem & 0xFFFFu) << 16;   // tensor_dim1[31:16]=0 | tile_dim0
  g1[4] = 0u;                        // tile_dim1=0, tile_dim2=0 (unused)
  g1[5] = nElem;                     // tensor_dim0_stride[31:0]
  g1[6] = (nElem & 0xFFFFu) << 16;   // stride0[47:32]=0 | stride1[15:0]
  g1[7] = 0u;                        // stride1[47:16]
  asm volatile("tensor_load_to_lds %0, %1" :: "s"(g0), "s"(g1) : "memory");
}

__global__ __launch_bounds__(kThreads)
void dpo_kernel(const float* __restrict__ X, float* __restrict__ O) {
  __shared__ __align__(16) float sXi[kT][kD];      // 8 KB  i-tile (resident)
  __shared__ __align__(16) float sXj[2][kT][kD];   // 16 KB ping-pong j-tiles
  __shared__ __align__(16) float sC[kT][kT];       // 1 KB  weight tile
  __shared__ float sS[kT];                         // row sums of C

  const int t    = threadIdx.x;
  const int lane = t & 31;
  const int wave = t >> 5;                      // 0..7 -> D chunk
  const int blk  = blockIdx.x;                  // 0..127
  const int b    = blk / (kL / kT);
  const int gi0  = (blk % (kL / kT)) * kT;

  const float* Xb = X + (size_t)b * kL * kD;
  const unsigned ldsXj0 = (unsigned)(unsigned long long)&sXj[0][0][0];
  const unsigned ldsXj1 = (unsigned)(unsigned long long)&sXj[1][0][0];

  // ---- stage resident Xi tile via CDNA5 async global->LDS (ASYNCcnt path) ----
  {
    const unsigned ldsBase = (unsigned)(unsigned long long)&sXi[0][0];
    const float4* src = (const float4*)(Xb + (size_t)gi0 * kD);
#pragma unroll
    for (int k = t; k < kT * kD / 4; k += kThreads) {
      unsigned long long ga = (unsigned long long)(src + k);
      unsigned ldsOff = ldsBase + (unsigned)k * 16u;
      asm volatile("global_load_async_to_lds_b128 %0, %1, off"
                   :: "v"(ldsOff), "v"(ga) : "memory");
    }
    asm volatile("s_wait_asynccnt 0x0" ::: "memory");
  }
  if (t < kT) sS[t] = 0.0f;

  // ---- TDM prologue: kick off tile 0 into buffer 0 ----
  if (wave == 0)
    tdm_load_tile(ldsXj0, (unsigned long long)Xb);
  __syncthreads();

  // each thread owns one (i,j) pair of the 16x16 distance tile
  const int ci = t >> 4;     // 0..15
  const int cj = t & 15;     // 0..15

  float s_part = 0.0f;
  v8f acc = {};              // C@X partial for this wave's D-chunk (f32)

  for (int jt = 0; jt < kNJT; ++jt) {
    const int cur = jt & 1;
    // ---- TDM: enqueue next tile into the other buffer, then wait for the
    //      current one. In-order TENSORcnt completion => tensorcnt<=1 means
    //      tile jt has landed while jt+1 still streams (latency hidden). ----
    if (wave == 0) {
      if (jt + 1 < kNJT) {
        tdm_load_tile(cur ? ldsXj0 : ldsXj1,
                      (unsigned long long)(Xb + (size_t)(jt + 1) * kT * kD));
        __builtin_amdgcn_s_wait_tensorcnt(1);
      } else {
        __builtin_amdgcn_s_wait_tensorcnt(0);
      }
    }
    // speculative prefetch two tiles ahead (global_prefetch_b8)
    if (jt + 2 < kNJT)
      __builtin_prefetch(Xb + (size_t)(jt + 2) * kT * kD + t * 8, 0, 1);
    __syncthreads();   // tile jt visible to all waves

    // ---- L1 distance + weight (VALU-bound phase) ----
    float d = 0.0f;
    const float4* xi = (const float4*)&sXi[ci][0];
    const float4* xj = (const float4*)&sXj[cur][cj][0];
#pragma unroll 8
    for (int k = 0; k < kD / 4; ++k) {
      float4 a = xi[k], q = xj[k];
      d += fabsf(a.x - q.x) + fabsf(a.y - q.y) +
           fabsf(a.z - q.z) + fabsf(a.w - q.w);
    }
    const int gi = gi0 + ci;
    const int gj = jt * kT + cj;
    const float alpha = 1.0f / (1.0f + fabsf((float)(gi - gj)));
    const float w     = 1.0f / (1.0f + d);
    const float c     = (gi == gj) ? 0.0f : alpha * w;
    sC[ci][cj] = c;
    s_part += c;
    __syncthreads();

    // ---- WMMA: acc += C_tile @ Xj_chunk  (K=16 zero-padded to 32) ----
    // A layout (16-bit 16x32): lanes 0-15 hold K=0..7, lanes 16-31 hold K=8..15
    v16h A = {};
    {
      const int mi = lane & 15;
      const int j0 = (lane >> 4) * 8;
#pragma unroll
      for (int h = 0; h < 8; ++h) A[h] = (_Float16)sC[mi][j0 + h];
    }
    // B layout (16-bit 32x16): lanes 0-15 hold K=0..7, lanes 16-31 hold K=8..15 of column N=lane&15
    v16h Bm = {};
    {
      const int col = wave * 16 + (lane & 15);
      const int k0 = (lane >> 4) * 8;
#pragma unroll
      for (int h = 0; h < 8; ++h) Bm[h] = (_Float16)sXj[cur][k0 + h][col];
    }
    acc = __builtin_amdgcn_wmma_f32_16x16x32_f16(
        /*neg_a=*/false, A, /*neg_b=*/false, Bm,
        /*c_mod=*/(short)0, acc, /*reuse_a=*/false, /*reuse_b=*/false);
    __syncthreads();   // protect sC and the buffer TDM will overwrite next
  }

  // ---- reduce row sums (ds_add_f32) ----
  atomicAdd(&sS[ci], s_part);
  __syncthreads();

  // ---- epilogue: O = sS[m]*Xi[m] - acc ; C/D layout: VGPR r -> M=r+8*(lane>=16), N=lane&15
  {
    const int n   = lane & 15;
    const int mhi = (lane >> 4) * 8;
    const int col = wave * 16 + n;
#pragma unroll
    for (int r = 0; r < 8; ++r) {
      const int m = mhi + r;
      const float val = sS[m] * sXi[m][col] - acc[r];
      O[((size_t)b * kL + (gi0 + m)) * kD + col] = val;
    }
  }
}

extern "C" void kernel_launch(void* const* d_in, const int* in_sizes, int n_in,
                              void* d_out, int out_size, void* d_ws, size_t ws_size,
                              hipStream_t stream) {
  (void)in_sizes; (void)n_in; (void)d_ws; (void)ws_size; (void)out_size;
  const float* X = (const float*)d_in[0];
  float* O = (float*)d_out;
  dim3 grid(kB * (kL / kT));   // 128 blocks: one per (batch, 16-row i-tile)
  dim3 block(kThreads);        // 8 wave32s
  hipLaunchKernelGGL(dpo_kernel, grid, block, 0, stream, X, O);
}
